// GCLSTM_64458869178653
// MI455X (gfx1250) — matrix operand order repaired and checked by reading, not compiled, not run on hardware
//
#include <hip/hip_runtime.h>

// Problem constants (fixed by the reference)
#define Bsz  512
#define Tt   24
#define NTm  10
#define FIN  256
#define FOUT 256
#define Nn   (Bsz*NTm)      // 5120 nodes
#define Ee   163840         // edges
#define KT   1024           // fused K = Fin + 3*Fout
#define GOUT 1024           // fused N = 4 gates * Fout

typedef __attribute__((ext_vector_type(8)))  float  v8f;
typedef __attribute__((ext_vector_type(8)))  __bf16 v8bf;
typedef __attribute__((ext_vector_type(16))) __bf16 v16bf;

// ---------------- CDNA5 async memory helpers ----------------
// GLOBAL_LOAD_ASYNC_TO_LDS_B128 (VGLOBAL op 98): 16B memory -> LDS, ASYNCcnt.
// vdst = VGPR with 32-bit LDS byte address, vaddr = 64-bit global address.
__device__ __forceinline__ void async_copy16(unsigned int lds_addr, const void* gptr) {
  asm volatile("global_load_async_to_lds_b128 %0, %1, off"
               :: "v"(lds_addr), "v"(gptr) : "memory");
}
__device__ __forceinline__ void wait_async4() {
#if defined(__has_builtin) && __has_builtin(__builtin_amdgcn_s_wait_asynccnt)
  __builtin_amdgcn_s_wait_asynccnt(4);
#else
  asm volatile("s_wait_asynccnt 0x4" ::: "memory");
#endif
}
__device__ __forceinline__ void wait_async0() {
#if defined(__has_builtin) && __has_builtin(__builtin_amdgcn_s_wait_asynccnt)
  __builtin_amdgcn_s_wait_asynccnt(0);
#else
  asm volatile("s_wait_asynccnt 0x0" ::: "memory");
#endif
}
__device__ __forceinline__ unsigned int lds_off(const void* p) {
  // generic "shared" address = {SHARED_BASE[63:32], lds_offset[31:0]}
  return (unsigned int)(unsigned long long)p;
}

// ---------------- utility kernels ----------------

__global__ void k_zero(float* __restrict__ p, long n) {
  long i = (long)blockIdx.x * blockDim.x + threadIdx.x;
  if (i < n) p[i] = 0.0f;
}

__global__ void k_neg(const float* __restrict__ a, float* __restrict__ o, long n) {
  long i = (long)blockIdx.x * blockDim.x + threadIdx.x;
  if (i < n) o[i] = -a[i];
}

// degree: deg[src] += w  (time-invariant)
__global__ void k_deg(const int* __restrict__ src, const float* __restrict__ w,
                      float* __restrict__ deg) {
  int e = blockIdx.x * blockDim.x + threadIdx.x;
  if (e < Ee) atomicAdd(&deg[src[e]], w[e]);
}

// w_hat = (2/lambda_max) * (-d^-1/2[src] * w * d^-1/2[dst]); 2/2.0 == 1
__global__ void k_what(const int* __restrict__ src, const int* __restrict__ dst,
                       const float* __restrict__ w, const float* __restrict__ deg,
                       float* __restrict__ what) {
  int e = blockIdx.x * blockDim.x + threadIdx.x;
  if (e >= Ee) return;
  float ds = deg[src[e]], dd = deg[dst[e]];
  float rs = ds > 0.0f ? rsqrtf(ds) : 0.0f;
  float rd = dd > 0.0f ? rsqrtf(dd) : 0.0f;
  what[e] = -(rs * w[e] * rd);
}

// Pack W_cat TRANSPOSED in bf16: Wt[n][k], n = g*256+c (gate column), k = fused K row.
// k-blocks: [0,256)=Ws[g], [256,512)=theta[g][0], [512,768)=theta[g][1], [768,1024)=theta[g][2]
__global__ void k_packw(const float* __restrict__ Ws, const float* __restrict__ thetas,
                        __bf16* __restrict__ Wt) {
  int idx = blockIdx.x * blockDim.x + threadIdx.x;   // over 1024*1024
  int kr = idx >> 10, gc = idx & 1023;
  int g = gc >> 8, c = gc & 255;
  int sel = kr >> 8, r = kr & 255;
  float v;
  if (sel == 0) v = Ws[(g * 256 + r) * 256 + c];
  else          v = thetas[((g * 3 + (sel - 1)) * 256 + r) * 256 + c];
  Wt[(size_t)gc * KT + kr] = (__bf16)v;
}

__global__ void k_bias(const float* __restrict__ bs, const float* __restrict__ cbs,
                       float* __restrict__ biasc) {
  int i = blockIdx.x * blockDim.x + threadIdx.x;
  if (i < GOUT) biasc[i] = bs[i] + cbs[i];
}

// out[dst] += scale * w_hat[e] * x[src]  (diag of L_hat is 0 for lambda_max=2)
__global__ __launch_bounds__(256) void k_scatter(
    const int* __restrict__ src, const int* __restrict__ dst,
    const float* __restrict__ what, const float* __restrict__ xin,
    float* __restrict__ out, float scale) {
  int wave = threadIdx.x >> 5, lane = threadIdx.x & 31;
  int e = blockIdx.x * 8 + wave;
  if (e >= Ee) return;
  float w = what[e] * scale;
  const float* xr = xin + (long)src[e] * FOUT;
  float* orow = out + (long)dst[e] * FOUT;
#pragma unroll
  for (int f = 0; f < FOUT; f += 32)
    atomicAdd(&orow[f + lane], w * xr[f + lane]);
}

// Zb row layout (bf16, KT wide): [X_t | H | Tx1 | Tx2]
__global__ void k_xcast(const float* __restrict__ X, int t, __bf16* __restrict__ Zb) {
  long idx = (long)blockIdx.x * blockDim.x + threadIdx.x;
  if (idx >= (long)Nn * FIN) return;
  int n = (int)(idx >> 8), f = (int)(idx & 255);
  Zb[(size_t)n * KT + f] = (__bf16)X[((size_t)n * Tt + t) * FIN + f];
}

__global__ void k_cvt(const float* __restrict__ Tx1, const float* __restrict__ Tx2,
                      __bf16* __restrict__ Zb) {
  long idx = (long)blockIdx.x * blockDim.x + threadIdx.x;
  if (idx >= (long)Nn * FOUT) return;
  int n = (int)(idx >> 8), f = (int)(idx & 255);
  Zb[(size_t)n * KT + 512 + f] = (__bf16)Tx1[idx];
  Zb[(size_t)n * KT + 768 + f] = (__bf16)Tx2[idx];
}

// ---------------- fused gate GEMM (WMMA bf16 / f32 acc, async double-buffer) ----
// gates(5120 x 1024) = Zb(5120 x 1024) @ Wt^T + bias
#define TM 128
#define TN 128
#define TK 32
#define LDP 40   // bf16 row padding: 80B rows => 16B-aligned segs, conflict-free frags

__global__ __launch_bounds__(256) void k_gemm(
    const __bf16* __restrict__ Zb, const __bf16* __restrict__ Wt,
    const float* __restrict__ biasc, float* __restrict__ gates) {
  __shared__ __align__(16) __bf16 As[2][TM][LDP];    // A tile (M x K), row-major
  __shared__ __align__(16) __bf16 Bst[2][TN][LDP];   // B tile (N x K), row-major

  const int m0 = blockIdx.x * TM;
  const int n0 = blockIdx.y * TN;
  const int tid  = threadIdx.x;
  const int lane = tid & 31;
  const int wave = tid >> 5;
  const int wm = (wave & 1) * 64;        // 2 wave rows x 64 M
  const int wn = (wave >> 1) * 32;       // 4 wave cols x 32 N
  const int r  = lane & 15;
  const int hi = lane >> 4;
  const int koffA = hi * 8;              // A frag: K = (e/8)*16 + koffA + e%8
  const int koffB = hi * 8;              // B frag: K = (e/8)*16 + koffB + e%8

  v8f acc[4][2] = {};

  // async tile copy: 128 rows x 64B each for A and B; 256 threads x 2 segs each
  const char* Zc = (const char*)Zb;
  const char* Wc = (const char*)Wt;
  const unsigned int ldsA0 = lds_off(&As[0][0][0]);
  const unsigned int ldsB0 = lds_off(&Bst[0][0][0]);
  const int crow = tid >> 2, cseg = tid & 3;   // rows 0..63 handled; +64 in 2nd pass

  auto issue = [&](int kb, int buf) {
    const unsigned int la = ldsA0 + (unsigned int)buf * (TM * LDP * 2);
    const unsigned int lb = ldsB0 + (unsigned int)buf * (TN * LDP * 2);
#pragma unroll
    for (int h = 0; h < 2; ++h) {
      const int rr = crow + h * 64;
      const unsigned int lo = (unsigned int)rr * (LDP * 2) + (unsigned int)cseg * 16;
      const size_t go = (size_t)cseg * 16 + (size_t)kb * 2;
      async_copy16(la + lo, Zc + (size_t)(m0 + rr) * (KT * 2) + go);
      async_copy16(lb + lo, Wc + (size_t)(n0 + rr) * (KT * 2) + go);
    }
  };

  issue(0, 0);
  for (int kb = 0, stage = 0; kb < KT; kb += TK, ++stage) {
    const int cur = stage & 1;
    if (kb + TK < KT) {
      issue(kb + TK, cur ^ 1);   // prefetch next tile into other buffer
      wait_async4();             // latest 4 still in flight; current tile landed
    } else {
      wait_async0();
    }
    __syncthreads();

    // B fragments (32x16 bf16): lane r holds N = wn+j*16+r, element e -> K = (e/8)*16 + koffB + e%8
    v16bf bfr[2];
#pragma unroll
    for (int j = 0; j < 2; ++j) {
      const v8bf* p = (const v8bf*)&Bst[cur][wn + j * 16 + r][koffB];
      v8bf l0 = p[0], h0 = p[2];
      bfr[j] = __builtin_shufflevector(l0, h0, 0,1,2,3,4,5,6,7,8,9,10,11,12,13,14,15);
    }
#pragma unroll
    for (int i = 0; i < 4; ++i) {
      // A fragment (16x32 bf16): element e -> K = (e/8)*16 + koffA + e%8
      const __bf16* ap = &As[cur][wm + i * 16 + r][0];
      v8bf l0 = *(const v8bf*)(ap + koffA);
      v8bf h0 = *(const v8bf*)(ap + 16 + koffA);
      v16bf afr = __builtin_shufflevector(l0, h0, 0,1,2,3,4,5,6,7,8,9,10,11,12,13,14,15);
#pragma unroll
      for (int j = 0; j < 2; ++j)
        acc[i][j] = __builtin_amdgcn_wmma_f32_16x16x32_bf16(
            false, afr, false, bfr[j], (short)0, acc[i][j], false, false);
    }
    __syncthreads();   // all waves done reading before next prefetch overwrites
  }

  // C/D layout: VGPR v, lanes 0-15 -> M=v, lanes 16-31 -> M=v+8; N = lane&15
  const int mrow = hi * 8;
#pragma unroll
  for (int i = 0; i < 4; ++i) {
    const int mbase = m0 + wm + i * 16 + mrow;
#pragma unroll
    for (int j = 0; j < 2; ++j) {
      const int nn = n0 + wn + j * 16 + r;
      const float bb = biasc[nn];
#pragma unroll
      for (int v = 0; v < 8; ++v)
        gates[(size_t)(mbase + v) * GOUT + nn] = acc[i][j][v] + bb;
    }
  }
}

// ---------------- fused LSTM pointwise update ----------------
__device__ __forceinline__ float sigf(float x) { return 1.0f / (1.0f + __expf(-x)); }

__global__ void k_lstm(const float* __restrict__ gates,
                       float* __restrict__ Hs, float* __restrict__ Cs,
                       __bf16* __restrict__ Zb) {
  long idx = (long)blockIdx.x * blockDim.x + threadIdx.x;
  if (idx >= (long)Nn * FOUT) return;
  int n = (int)(idx >> 8), f = (int)(idx & 255);
  const float* g = gates + (size_t)n * GOUT;
  float I  = sigf(g[f]);
  float Fg = sigf(g[256 + f]);
  float Tc = tanhf(g[512 + f]);
  float O  = sigf(g[768 + f]);
  float c  = Fg * Cs[idx] + I * Tc;
  float h  = O * tanhf(c);
  Cs[idx] = c;
  Hs[idx] = h;
  Zb[(size_t)n * KT + 256 + f] = (__bf16)h;   // bf16 H for next step's GEMM
}

// ---------------- host orchestration ----------------
extern "C" void kernel_launch(void* const* d_in, const int* in_sizes, int n_in,
                              void* d_out, int out_size, void* d_ws, size_t ws_size,
                              hipStream_t stream) {
  const float* X      = (const float*)d_in[0];
  const float* ew     = (const float*)d_in[1];
  const float* Ws     = (const float*)d_in[2];
  const float* bs     = (const float*)d_in[3];
  const float* thetas = (const float*)d_in[4];
  const float* cbs    = (const float*)d_in[5];
  const int*   ei     = (const int*)d_in[6];
  const int* esrc = ei;
  const int* edst = ei + Ee;

  char* ws = (char*)d_ws;
  float*  what  = (float*)ws;  ws += (size_t)Ee * 4;
  float*  deg   = (float*)ws;  ws += (size_t)Nn * 4;
  float*  Tx1   = (float*)ws;  ws += (size_t)Nn * FOUT * 4;
  float*  Tx2   = (float*)ws;  ws += (size_t)Nn * FOUT * 4;
  __bf16* Zb    = (__bf16*)ws; ws += (size_t)Nn * KT * 2;
  __bf16* Wt    = (__bf16*)ws; ws += (size_t)KT * GOUT * 2;
  float*  biasc = (float*)ws;  ws += (size_t)GOUT * 4;
  float*  gates = (float*)ws;  ws += (size_t)Nn * GOUT * 4;

  float* H = (float*)d_out;                // output: H then C, concatenated
  float* C = H + (size_t)Nn * FOUT;

  const long NF = (long)Nn * FOUT;
  dim3 blk(256);

  // one-time (per call) preprocessing
  k_zero<<<dim3((Nn + 255) / 256), blk, 0, stream>>>(deg, Nn);
  k_deg<<<dim3((Ee + 255) / 256), blk, 0, stream>>>(esrc, ew, deg);
  k_what<<<dim3((Ee + 255) / 256), blk, 0, stream>>>(esrc, edst, ew, deg, what);
  k_packw<<<dim3((KT * GOUT) / 256), blk, 0, stream>>>(Ws, thetas, Wt);
  k_bias<<<dim3(4), blk, 0, stream>>>(bs, cbs, biasc);
  k_zero<<<dim3((unsigned)((2 * NF + 255) / 256)), blk, 0, stream>>>(H, 2 * NF);
  // zero Zb (bf16, zero bit pattern) -> H block starts at 0 for t=0
  k_zero<<<dim3((unsigned)(((long)Nn * KT / 2 + 255) / 256)), blk, 0, stream>>>(
      (float*)Zb, (long)Nn * KT / 2);

  dim3 gemm_grid(Nn / TM, GOUT / TN);   // 40 x 8
  for (int t = 0; t < Tt; ++t) {
    // Tx1 = L_hat @ H  (diag term is exactly 0)
    k_zero<<<dim3((unsigned)((NF + 255) / 256)), blk, 0, stream>>>(Tx1, NF);
    k_scatter<<<dim3(Ee / 8), blk, 0, stream>>>(esrc, edst, what, H, Tx1, 1.0f);
    // Tx2 = 2 * L_hat @ Tx1 - H
    k_neg<<<dim3((unsigned)((NF + 255) / 256)), blk, 0, stream>>>(H, Tx2, NF);
    k_scatter<<<dim3(Ee / 8), blk, 0, stream>>>(esrc, edst, what, Tx1, Tx2, 2.0f);
    // assemble bf16 activation rows [X_t | H | Tx1 | Tx2]
    k_xcast<<<dim3((unsigned)((NF + 255) / 256)), blk, 0, stream>>>(X, t, Zb);
    k_cvt<<<dim3((unsigned)((NF + 255) / 256)), blk, 0, stream>>>(Tx1, Tx2, Zb);
    // fused 4-gate GEMM
    k_gemm<<<gemm_grid, blk, 0, stream>>>(Zb, Wt, biasc, gates);
    // LSTM pointwise update (H, C in d_out; also emits bf16 H for next step)
    k_lstm<<<dim3((unsigned)((NF + 255) / 256)), blk, 0, stream>>>(gates, H, C, Zb);
  }
}
